// MultiHeadAttention_69200513073169
// MI455X (gfx1250) — matrix operand order repaired and checked by reading, not compiled, not run on hardware
//
#include <hip/hip_runtime.h>

#define DM 1024
#define NH 16
#define DK 64
#define SEQ 2048
#define QKV_LD 3072

typedef __attribute__((ext_vector_type(16))) _Float16 v16h;
typedef __attribute__((ext_vector_type(8)))  _Float16 v8h;
typedef __attribute__((ext_vector_type(4)))  _Float16 v4h;
typedef __attribute__((ext_vector_type(8)))  float    v8f;

union FragA { v16h v; v4h q[4]; };
union FragB { v16h v; v8h o[2]; };
union FragC { v8f  v; float f[8]; };

__device__ __forceinline__ v8f wmma_f16(v16h a, v16h b, v8f c) {
  // D = A(16x32 f16) * B(32x16 f16) + C(16x16 f32)
  return __builtin_amdgcn_wmma_f32_16x16x32_f16(false, a, false, b, (short)0, c,
                                                false, false);
}

// CDNA5 async global->LDS copy (ASYNCcnt-tracked, no VGPR round trip).
// vdst = per-lane LDS byte address, vaddr = 64-bit global address.
__device__ __forceinline__ void async_ld_b128(unsigned lds_off, const void* g) {
  asm volatile("global_load_async_to_lds_b128 %0, %1, off"
               :: "v"(lds_off), "v"((unsigned long long)(size_t)g)
               : "memory");
}
__device__ __forceinline__ void wait_async0() {
  asm volatile("s_wait_asynccnt 0x0" ::: "memory");
}

// ---------------------------------------------------------------------------
// fp32 -> f16 convert (vectorized float4 -> 4 halves)
// ---------------------------------------------------------------------------
__global__ __launch_bounds__(256) void k_cvt_f16(const float* __restrict__ in,
                                                 _Float16* __restrict__ out,
                                                 int n4) {
  int i = blockIdx.x * 256 + threadIdx.x;
  if (i >= n4) return;
  const float4 f = ((const float4*)in)[i];
  v4h h4;
  h4[0] = (_Float16)f.x; h4[1] = (_Float16)f.y;
  h4[2] = (_Float16)f.z; h4[3] = (_Float16)f.w;
  ((v4h*)out)[i] = h4;
}

// ---------------------------------------------------------------------------
// W [K][N] fp32 -> Wt [N][K] f16 (one-time; L2-resident, coalesced writes)
// ---------------------------------------------------------------------------
__global__ __launch_bounds__(256) void k_transpose_f16(const float* __restrict__ in,
                                                       _Float16* __restrict__ out,
                                                       int K, int N) {
  int idx = blockIdx.x * 256 + threadIdx.x;
  if (idx >= K * N) return;
  int n = idx / K;
  int k = idx - n * K;
  out[idx] = (_Float16)in[(size_t)k * N + n];
}

// ---------------------------------------------------------------------------
// GEMM: C[M,N] = A[M,K](f16,row) * Bt[N,K](f16,row) + bias[N]
// block tile 128x128, 8 waves (4x2), wave tile 32x64, K-step 32.
// Double-buffered async global->LDS pipeline: DMA of tile i+1 overlaps
// WMMA compute of tile i; one barrier + one s_wait_asynccnt per K-step.
// ---------------------------------------------------------------------------
template <bool OUT_F16>
__global__ __launch_bounds__(256) void k_gemm(const _Float16* __restrict__ A,
                                              const _Float16* __restrict__ Bt,
                                              const float* __restrict__ bias,
                                              void* __restrict__ Cout,
                                              int M, int N, int K) {
  constexpr int LDA = 40;  // 32 + 8 halves pad -> 80B row stride, conflict-free
  __shared__ _Float16 smem[2][2][128 * LDA];  // [buf][A/B][tile]

  const int t = threadIdx.x;
  const int lane = t & 31, wave = t >> 5;
  const int half = lane >> 4, ln = lane & 15;
  const int wm = wave >> 1, wn = wave & 1;
  const int bm = blockIdx.y * 128, bn = blockIdx.x * 128;

  FragC acc[2][4];
#pragma unroll
  for (int mt = 0; mt < 2; ++mt)
#pragma unroll
    for (int nt = 0; nt < 4; ++nt)
#pragma unroll
      for (int r = 0; r < 8; ++r) acc[mt][nt].f[r] = 0.f;

  // each thread DMAs 32B of A and 32B of B per K-step
  const int grow = t >> 1;
  const int gseg = (t & 1) * 16;
  const _Float16* gA = A + (size_t)(bm + grow) * K + gseg;
  const _Float16* gB = Bt + (size_t)(bn + grow) * K + gseg;
  const unsigned lbase = (unsigned)(size_t)&smem[0][0][0];
  const unsigned aoff = lbase + (unsigned)((grow * LDA + gseg) * 2);
  const unsigned boff = aoff + (unsigned)(128 * LDA * 2);
  const unsigned bufstride = (unsigned)(2 * 128 * LDA * 2);

  // prologue: DMA tile 0 into buffer 0
  {
    async_ld_b128(aoff, gA);
    async_ld_b128(aoff + 16, gA + 8);
    async_ld_b128(boff, gB);
    async_ld_b128(boff + 16, gB + 8);
  }

  for (int k0 = 0, it = 0; k0 < K; k0 += 32, ++it) {
    const int buf = it & 1;
    wait_async0();      // this wave's DMA for tile `it` has landed in LDS
    __syncthreads();    // every wave's DMA landed; prior-buffer reads done

    if (k0 + 32 < K) {  // kick off DMA for the next tile into the other buffer
      const unsigned d = (buf ^ 1) ? bufstride : 0u;
      const _Float16* pa = gA + (k0 + 32);
      const _Float16* pb = gB + (k0 + 32);
      async_ld_b128(aoff + d, pa);
      async_ld_b128(aoff + d + 16, pa + 8);
      async_ld_b128(boff + d, pb);
      async_ld_b128(boff + d + 16, pb + 8);
    }

    const _Float16* sAb = smem[buf][0];
    const _Float16* sBb = smem[buf][1];

    FragA fa[2];
#pragma unroll
    for (int mt = 0; mt < 2; ++mt) {
      const _Float16* p = &sAb[(wm * 32 + mt * 16 + ln) * LDA];
      fa[mt].q[0] = *(const v4h*)(p + 16 * half);
      fa[mt].q[1] = *(const v4h*)(p + 16 * half + 4);
      fa[mt].q[2] = *(const v4h*)(p + 16 * half + 8);
      fa[mt].q[3] = *(const v4h*)(p + 16 * half + 12);
    }
    FragB fb[4];
#pragma unroll
    for (int nt = 0; nt < 4; ++nt) {
      const _Float16* p = &sBb[(wn * 64 + nt * 16 + ln) * LDA + 16 * half];
      fb[nt].o[0] = *(const v8h*)p;
      fb[nt].o[1] = *(const v8h*)(p + 8);
    }
#pragma unroll
    for (int mt = 0; mt < 2; ++mt)
#pragma unroll
      for (int nt = 0; nt < 4; ++nt)
        acc[mt][nt].v = wmma_f16(fa[mt].v, fb[nt].v, acc[mt][nt].v);
  }

#pragma unroll
  for (int nt = 0; nt < 4; ++nt) {
    const int col = bn + wn * 64 + nt * 16 + ln;
    const float bv = bias[col];
#pragma unroll
    for (int mt = 0; mt < 2; ++mt) {
#pragma unroll
      for (int r = 0; r < 8; ++r) {
        const int rowg = bm + wm * 32 + mt * 16 + r + 8 * half;
        const float vv = acc[mt][nt].f[r] + bv;
        if (OUT_F16)
          ((_Float16*)Cout)[(size_t)rowg * N + col] = (_Float16)vv;
        else
          ((float*)Cout)[(size_t)rowg * N + col] = vv;
      }
    }
  }
}

// ---------------------------------------------------------------------------
// Flash attention (causal). One block = (b, h, 128 q-rows); 8 waves,
// each wave owns 16 q-rows. Key chunks of 64, online softmax.
// Scale 1/sqrt(dk) folded into Q (exact pow2); causal mask applied only on
// the single diagonal chunk per wave — interior chunks are branch-free.
// qkv: f16 [B*S][3072]  (Q at +0, K at +1024, V at +2048, head h at h*64)
// outh: f16 [B*S][1024] merged-head layout.
// ---------------------------------------------------------------------------
__global__ __launch_bounds__(256) void k_flash(const _Float16* __restrict__ qkv,
                                               _Float16* __restrict__ outh) {
  constexpr int LVK = 72;                 // padded key stride (halves)
  __shared__ _Float16 sVT[64 * LVK];      // V^T chunk: [d][key]
  __shared__ _Float16 sP[8 * 16 * LVK];   // per-wave P staging [m][key]

  const int t = threadIdx.x;
  const int lane = t & 31, wave = t >> 5;
  const int half = lane >> 4, ln = lane & 15;
  const int qb = blockIdx.x & 15;
  const int h = (blockIdx.x >> 4) & 15;
  const int b = blockIdx.x >> 8;
  const int q0 = qb * 128 + wave * 16;
  const size_t bh = (size_t)b * SEQ;

  // Q fragments (16 rows x 64 d); fold 1/sqrt(64)=0.125 (exact in f16)
  FragA aq[2];
  {
    const _Float16* qp = qkv + (bh + q0 + ln) * QKV_LD + h * DK;
#pragma unroll
    for (int dh = 0; dh < 2; ++dh) {
      const _Float16* p = qp + dh * 32;
      aq[dh].q[0] = *(const v4h*)(p + 16 * half);
      aq[dh].q[1] = *(const v4h*)(p + 16 * half + 4);
      aq[dh].q[2] = *(const v4h*)(p + 16 * half + 8);
      aq[dh].q[3] = *(const v4h*)(p + 16 * half + 12);
#pragma unroll
      for (int i = 0; i < 16; ++i)
        aq[dh].v[i] = aq[dh].v[i] * (_Float16)0.125f;
    }
  }

  float m_i[8], l_i[8];
  FragC acc[4];
#pragma unroll
  for (int r = 0; r < 8; ++r) { m_i[r] = -1e30f; l_i[r] = 0.f; }
#pragma unroll
  for (int nt = 0; nt < 4; ++nt)
#pragma unroll
    for (int r = 0; r < 8; ++r) acc[nt].f[r] = 0.f;

  const int vkey = t >> 2;       // 0..63 key within chunk
  const int vd0 = (t & 3) * 16;  // 0,16,32,48 d offset
  const int jmax = qb * 128 + 128;

  for (int j0 = 0; j0 < jmax; j0 += 64) {
    __syncthreads();  // protect previous iteration's sVT reads
    {
      const _Float16* vp =
          qkv + (bh + j0 + vkey) * QKV_LD + 2 * DM + h * DK + vd0;
      const v8h v0 = *(const v8h*)vp;
      const v8h v1 = *(const v8h*)(vp + 8);
#pragma unroll
      for (int i = 0; i < 8; ++i) sVT[(vd0 + i) * LVK + vkey] = v0[i];
#pragma unroll
      for (int i = 0; i < 8; ++i) sVT[(vd0 + 8 + i) * LVK + vkey] = v1[i];
    }
    __syncthreads();

    if (j0 > q0 + 15) continue;  // fully-masked chunk for this wave

    // scores S = (Q/8) * K^T  (16 x 64)
    FragC sc[4];
#pragma unroll
    for (int nt = 0; nt < 4; ++nt) {
#pragma unroll
      for (int r = 0; r < 8; ++r) sc[nt].f[r] = 0.f;
      FragB bk;
#pragma unroll
      for (int dh = 0; dh < 2; ++dh) {
        const _Float16* kp = qkv + (bh + j0 + nt * 16 + ln) * QKV_LD + DM +
                             h * DK + dh * 32 + 16 * half;
        bk.o[0] = *(const v8h*)kp;
        bk.o[1] = *(const v8h*)(kp + 8);
        sc[nt].v = wmma_f16(aq[dh].v, bk.v, sc[nt].v);
      }
    }
    // causal mask only on the (single) diagonal chunk of this wave
    if (j0 + 63 > q0) {
#pragma unroll
      for (int nt = 0; nt < 4; ++nt) {
        const int key = j0 + nt * 16 + ln;
#pragma unroll
        for (int r = 0; r < 8; ++r) {
          const int qr = q0 + r + 8 * half;
          if (key > qr) sc[nt].f[r] = -1e30f;
        }
      }
    }
    // row max (butterfly within 16-lane half owning the row group)
    float mx[8];
#pragma unroll
    for (int r = 0; r < 8; ++r)
      mx[r] = fmaxf(fmaxf(sc[0].f[r], sc[1].f[r]),
                    fmaxf(sc[2].f[r], sc[3].f[r]));
#pragma unroll
    for (int off = 1; off < 16; off <<= 1)
#pragma unroll
      for (int r = 0; r < 8; ++r)
        mx[r] = fmaxf(mx[r], __shfl_xor(mx[r], off, 32));
    float corr[8];
#pragma unroll
    for (int r = 0; r < 8; ++r) {
      const float mn = fmaxf(m_i[r], mx[r]);
      corr[r] = __expf(m_i[r] - mn);
      m_i[r] = mn;
    }
    // p = exp(s - m)
#pragma unroll
    for (int nt = 0; nt < 4; ++nt)
#pragma unroll
      for (int r = 0; r < 8; ++r) sc[nt].f[r] = __expf(sc[nt].f[r] - m_i[r]);
    float rs[8];
#pragma unroll
    for (int r = 0; r < 8; ++r)
      rs[r] = (sc[0].f[r] + sc[1].f[r]) + (sc[2].f[r] + sc[3].f[r]);
#pragma unroll
    for (int off = 1; off < 16; off <<= 1)
#pragma unroll
      for (int r = 0; r < 8; ++r) rs[r] += __shfl_xor(rs[r], off, 32);
#pragma unroll
    for (int r = 0; r < 8; ++r) l_i[r] = l_i[r] * corr[r] + rs[r];
#pragma unroll
    for (int nt = 0; nt < 4; ++nt)
#pragma unroll
      for (int r = 0; r < 8; ++r) acc[nt].f[r] *= corr[r];

    // C-layout -> A-layout: bounce P through wave-private LDS (f16)
    _Float16* pw = &sP[wave * 16 * LVK];
#pragma unroll
    for (int nt = 0; nt < 4; ++nt)
#pragma unroll
      for (int r = 0; r < 8; ++r)
        pw[(r + 8 * half) * LVK + nt * 16 + ln] = (_Float16)sc[nt].f[r];

    FragA pa[2];
#pragma unroll
    for (int kk = 0; kk < 2; ++kk) {
      const _Float16* p = &pw[ln * LVK + kk * 32];
      pa[kk].q[0] = *(const v4h*)(p + 16 * half);
      pa[kk].q[1] = *(const v4h*)(p + 16 * half + 4);
      pa[kk].q[2] = *(const v4h*)(p + 16 * half + 8);
      pa[kk].q[3] = *(const v4h*)(p + 16 * half + 12);
    }
    // acc += P(16x64) * V(64x64)
#pragma unroll
    for (int kk = 0; kk < 2; ++kk)
#pragma unroll
      for (int nt = 0; nt < 4; ++nt) {
        FragB bv;
        const _Float16* p = &sVT[(nt * 16 + ln) * LVK + kk * 32 + 16 * half];
        bv.o[0] = *(const v8h*)p;
        bv.o[1] = *(const v8h*)(p + 8);
        acc[nt].v = wmma_f16(pa[kk].v, bv.v, acc[nt].v);
      }
  }

  // epilogue: normalize by l and write merged-head f16 [B*S][DM]
#pragma unroll
  for (int r = 0; r < 8; ++r) {
    const float inv = 1.0f / l_i[r];
    const size_t row = bh + q0 + r + 8 * half;
#pragma unroll
    for (int nt = 0; nt < 4; ++nt)
      outh[row * DM + h * DK + nt * 16 + ln] =
          (_Float16)(acc[nt].f[r] * inv);
  }
}

// ---------------------------------------------------------------------------
extern "C" void kernel_launch(void* const* d_in, const int* in_sizes, int n_in,
                              void* d_out, int out_size, void* d_ws,
                              size_t ws_size, hipStream_t stream) {
  (void)n_in; (void)out_size; (void)ws_size;
  const float* x    = (const float*)d_in[0];
  const float* Wqkv = (const float*)d_in[1];
  const float* bqkv = (const float*)d_in[2];
  const float* Wo   = (const float*)d_in[3];
  const float* bo   = (const float*)d_in[4];

  const int BS = in_sizes[0] / DM;  // B*S (= 8192)
  const int B = BS / SEQ;           // 4

  char* ws = (char*)d_ws;
  size_t off = 0;
  _Float16* xh = (_Float16*)(ws + off);     off += (size_t)BS * DM * 2;
  _Float16* wqkvt = (_Float16*)(ws + off);  off += (size_t)3 * DM * DM * 2;
  _Float16* wot = (_Float16*)(ws + off);    off += (size_t)DM * DM * 2;
  _Float16* qkv = (_Float16*)(ws + off);    off += (size_t)BS * 3 * DM * 2;
  _Float16* attnh = (_Float16*)(ws + off);

  // 1) convert x -> f16
  {
    int n4 = BS * DM / 4;
    k_cvt_f16<<<(n4 + 255) / 256, 256, 0, stream>>>(x, xh, n4);
  }
  // 2) transpose-convert weights -> [N][K] f16
  {
    int n = 3 * DM * DM;
    k_transpose_f16<<<(n + 255) / 256, 256, 0, stream>>>(Wqkv, wqkvt, DM, 3 * DM);
  }
  {
    int n = DM * DM;
    k_transpose_f16<<<(n + 255) / 256, 256, 0, stream>>>(Wo, wot, DM, DM);
  }
  // 3) QKV projection: [BS,1024] x [1024,3072] + b_qkv -> f16 qkv
  {
    dim3 g(3 * DM / 128, BS / 128);
    k_gemm<true><<<g, 256, 0, stream>>>(xh, wqkvt, bqkv, qkv, BS, 3 * DM, DM);
  }
  // 4) causal flash attention per (b, h, 128-row q block)
  {
    k_flash<<<B * NH * (SEQ / 128), 256, 0, stream>>>(qkv, attnh);
  }
  // 5) output projection: [BS,1024] x [1024,1024] + b_o -> fp32 d_out
  {
    dim3 g(DM / 128, BS / 128);
    k_gemm<false><<<g, 256, 0, stream>>>(attnh, wot, bo, (float*)d_out, BS, DM, DM);
  }
}
